// STGCN_44023414784018
// MI455X (gfx1250) — matrix operand order repaired and checked by reading, not compiled, not run on hardware
//
#include <hip/hip_runtime.h>
#include <hip/hip_bf16.h>
#include <stdint.h>

// ---------------- problem constants ----------------
constexpr int Bn   = 8;
constexpr int Nn   = 2048;
constexpr int Ln   = 8;
constexpr int Cc   = 64;
constexpr int NNZn = 131072;
constexpr int CIN0 = 128;    // C + ADIM
constexpr int L1   = 7;      // after conv1 (K=2)
constexpr int L2c  = 6;      // after conv2 (K=2)
constexpr int XPK_BATCH = 64 * 28 * 512;   // halves per batch of packed X

typedef unsigned short u16;
typedef __bf16 bf16v16 __attribute__((ext_vector_type(16)));
typedef float  f32v8   __attribute__((ext_vector_type(8)));
typedef int    v4i     __attribute__((vector_size(16)));   // matches builtin param type

struct B32x8 { uint4 lo, hi; };

#define DEVFN __device__ __forceinline__

DEVFN u16 f2bf(float f) {
  uint32_t u = __builtin_bit_cast(uint32_t, f);
  u += 0x7FFFu + ((u >> 16) & 1u);          // round-to-nearest-even
  return (u16)(u >> 16);
}
DEVFN float bf2f(u16 h) {
  uint32_t u = ((uint32_t)h) << 16;
  return __builtin_bit_cast(float, u);
}
// A fragment (16x32 bf16, row-major source): lane already offset to its row/kb.
DEVFN bf16v16 load_frag_a(const u16* p) {
  B32x8 r{ *(const uint4*)p, *(const uint4*)(p + 16) };
  return __builtin_bit_cast(bf16v16, r);
}
// B fragment from packed buffer (16 contiguous halves per lane, global or LDS).
DEVFN bf16v16 load_frag_b(const u16* p) {
  B32x8 r{ *(const uint4*)p, *(const uint4*)(p + 8) };
  return __builtin_bit_cast(bf16v16, r);
}
DEVFN f32v8 wmma_bf16(bf16v16 a, bf16v16 b, f32v8 c) {
  return __builtin_amdgcn_wmma_f32_16x16x32_bf16(false, a, false, b, (short)0, c,
                                                 false, false);
}

// -------- global -> LDS copy: async path if toolchain exposes it -----
#if __has_builtin(__builtin_amdgcn_global_load_async_to_lds_b128)
#define HAS_ASYNC_LDS 1
DEVFN void cp_g2s_b128(u16* ldst, const u16* gsrc) {
  __builtin_amdgcn_global_load_async_to_lds_b128(
      (__attribute__((address_space(1))) v4i*)gsrc,
      (__attribute__((address_space(3))) v4i*)ldst, 0, 0);
}
#else
#define HAS_ASYNC_LDS 0
DEVFN void cp_g2s_b128(u16* ldst, const u16* gsrc) {
  *(uint4*)ldst = *(const uint4*)gsrc;      // global_load_b128 + ds_store_b128
}
#endif
DEVFN void wait_cp_all() {
#if HAS_ASYNC_LDS
#if __has_builtin(__builtin_amdgcn_s_wait_asynccnt)
  __builtin_amdgcn_s_wait_asynccnt(0);
#else
  asm volatile("s_wait_asynccnt 0x0" ::: "memory");
#endif
#endif
}

// ---------------- small prep kernels ----------------
__global__ void deg_init_k(float* deg, int n) {
  int i = blockIdx.x * blockDim.x + threadIdx.x;
  if (i < n) deg[i] = 1.0f;                 // identity contribution
}

__global__ void deg_scatter_k(const float* __restrict__ A, float* __restrict__ deg) {
  int i = blockIdx.x * blockDim.x + threadIdx.x;
  if (i >= Bn * NNZn) return;
  int b = i / NNZn, e = i % NNZn;
  const float* base = A + (size_t)b * 3 * NNZn;
  float v = base[e];
  float rf = base[NNZn + e];
  float cf = base[2 * NNZn + e];
  if (cf < (float)(Ln * Nn))
    atomicAdd(&deg[b * Nn + (int)rf], v);
}

__global__ void dinv_k(float* deg, int n) {
  int i = blockIdx.x * blockDim.x + threadIdx.x;
  if (i < n) deg[i] = rsqrtf(deg[i]);
}

__global__ void zero_f4_k(float4* p, size_t n4) {
  size_t i = (size_t)blockIdx.x * blockDim.x + threadIdx.x;
  if (i < n4) p[i] = make_float4(0.f, 0.f, 0.f, 0.f);
}

__global__ void adj_scatter_k(const float* __restrict__ A, float* __restrict__ Acnt) {
  int i = blockIdx.x * blockDim.x + threadIdx.x;
  if (i >= Bn * NNZn) return;
  int b = i / NNZn, e = i % NNZn;
  const float* base = A + (size_t)b * 3 * NNZn;
  float v = base[e];
  float rf = base[NNZn + e];
  float cf = base[2 * NNZn + e];
  if (cf < (float)(Ln * Nn)) {
    int r = (int)rf;
    int c = ((int)cf) % Nn;
    atomicAdd(&Acnt[((size_t)b * Nn + r) * Nn + c], v);
  }
}

// vectorized: 4 elements/thread (float4 in, ushort4 out) — pure bandwidth kernel
__global__ void adj_norm_k(const float* __restrict__ Acnt, const float* __restrict__ dinv,
                           u16* __restrict__ Abf) {
  size_t i = (size_t)blockIdx.x * blockDim.x + threadIdx.x;
  size_t total4 = (size_t)Bn * Nn * Nn / 4;
  if (i >= total4) return;
  size_t e0 = i * 4;
  int n0 = (int)(e0 % Nn);
  size_t bm = e0 / Nn;
  int m = (int)(bm % Nn);
  int b = (int)(bm / Nn);
  float4 v = ((const float4*)Acnt)[i];
  float dm = dinv[b * Nn + m];
  const float* dn = dinv + b * Nn + n0;
  float vv[4] = {v.x, v.y, v.z, v.w};
  ushort4 o;
  u16 r[4];
#pragma unroll
  for (int j = 0; j < 4; ++j)
    r[j] = f2bf((vv[j] + ((m == n0 + j) ? 1.0f : 0.0f)) * dm * dn[j]);
  o.x = r[0]; o.y = r[1]; o.z = r[2]; o.w = r[3];
  ((ushort4*)Abf)[i] = o;
}

__global__ void concat_bf16_k(const float* __restrict__ ps, const float* __restrict__ an,
                              u16* __restrict__ xbf) {
  int i = blockIdx.x * blockDim.x + threadIdx.x;       // over total/4
  if (i >= Bn * Nn * Ln * CIN0 / 4) return;
  int e0 = i * 4;
  int ci = e0 & (CIN0 - 1);
  int rest = e0 >> 7;
  const float* s = (ci < Cc) ? (ps + (size_t)rest * Cc + ci)
                             : (an + (size_t)rest * Cc + (ci - Cc));
  float4 v = *(const float4*)s;
  ushort4 o;
  o.x = f2bf(v.x); o.y = f2bf(v.y); o.z = f2bf(v.z); o.w = f2bf(v.w);
  ((ushort4*)xbf)[i] = o;
}

// Pack a row-major (Kd x Nd) f32 matrix into WMMA B-fragment layout (bf16).
__global__ void pack_b_f32_k(const float* __restrict__ src, u16* __restrict__ dst,
                             int Kd, int Nd) {
  int i = blockIdx.x * blockDim.x + threadIdx.x;
  if (i >= Kd * Nd) return;
  int tile = i >> 9, w = i & 511;
  int lane = w >> 4, h = w & 15;
  int ntiles = Nd >> 4;
  int kt = tile / ntiles, nt = tile % ntiles;
  int k = kt * 32 + ((h & 8) << 1) + ((lane >> 4) << 3) + (h & 7);
  int n = nt * 16 + (lane & 15);
  dst[i] = f2bf(src[(size_t)k * Nd + n]);
}

// ---------------- WMMA GEMM kernels ----------------
// Gated temporal conv as GEMM (templated on k-tile count -> fully unrolled,
// no A-register rotation copies). Weight B-fragments staged in LDS per block.
// packMode=1: epilogue writes directly in graph-GEMM B-fragment layout.
template <int KT>
__global__ void __launch_bounds__(256)
conv_glu_wmma_k(const u16* __restrict__ X, int xRowStride, int tStride,
                const u16* __restrict__ Wp, const u16* __restrict__ Wq,
                const float* __restrict__ bp, const float* __restrict__ bq,
                u16* __restrict__ Y, int yRowStride, int Lout, int packMode,
                int nWaves) {
  extern __shared__ u16 smem[];
  constexpr int Kf = KT * 32;
  constexpr int packHalfs = Kf * 64;              // per matrix
  int tid = threadIdx.x;
  constexpr int n4 = (packHalfs * 2) >> 3;        // uint4 copies (p then q)
  for (int i = tid; i < n4; i += 256) {
    int hoff = i * 8;
    const u16* src = (hoff < packHalfs) ? (Wp + hoff) : (Wq + (hoff - packHalfs));
    cp_g2s_b128(smem + hoff, src);
  }
  wait_cp_all();
  __syncthreads();

  int wave = (blockIdx.x * 256 + tid) >> 5;
  if (wave >= nWaves) return;                     // wave-uniform guard
  int mt = wave / Lout;
  int t  = wave % Lout;
  int m0 = mt * 16;
  int lane = tid & 31;
  int kb = (lane >> 4) << 3;                      // 0 or 8
  const u16* arow = X + (size_t)(m0 + (lane & 15)) * xRowStride + t * tStride + kb;
  const u16* wpl = smem + lane * 16;
  const u16* wql = smem + packHalfs + lane * 16;

  f32v8 accP[4] = {}, accQ[4] = {};
  bf16v16 a = load_frag_a(arow);                  // pipelined A stream
#pragma unroll
  for (int kt = 0; kt < KT; ++kt) {
    bf16v16 ac = a;                               // SSA-renamed away when unrolled
    if (kt + 1 < KT) a = load_frag_a(arow + (kt + 1) * 32);
#pragma unroll
    for (int nt = 0; nt < 4; ++nt) {
      accP[nt] = wmma_bf16(ac, load_frag_b(wpl + (kt * 4 + nt) * 512), accP[nt]);
      accQ[nt] = wmma_bf16(ac, load_frag_b(wql + (kt * 4 + nt) * 512), accQ[nt]);
    }
  }
  int row0 = m0 + ((lane >> 4) << 3);
  int colb = (lane & 15);
#pragma unroll
  for (int nt = 0; nt < 4; ++nt) {
    int c = nt * 16 + colb;                       // output channel
    int col = t * 64 + c;                         // column in (l,c) space
    float vbp = bp[c], vbq = bq[c];
#pragma unroll
    for (int r = 0; r < 8; ++r) {
      int row = row0 + r;
      float p = accP[nt][r] + vbp;
      float q = accQ[nt][r] + vbq;
      float y = p / (1.0f + __expf(-q));          // GLU
      if (packMode) {                             // write graph-GEMM B-fragment layout
        int rb = row & (Nn - 1), bb = row >> 11;
        int kt2 = rb >> 5, hi = (rb >> 3) & 1, h = ((rb >> 1) & 8) | (rb & 7);
        int nt2 = col >> 4, lo = col & 15;
        Y[(size_t)bb * XPK_BATCH +
          (size_t)((kt2 * 28 + nt2) * 32 + hi * 16 + lo) * 16 + h] = f2bf(y);
      } else {
        Y[(size_t)row * yRowStride + col] = f2bf(y);
      }
    }
  }
}

// Graph conv: per batch C(2048x448) = A_hat(2048x2048) x X(2048x448).
// Block = 8 waves sharing one (batch, 64-col tile); each wave owns a 32x64
// output tile (two A-row fragments share every B fragment -> 2 WMMA per
// ds-loaded fragment). X k-chunks double-buffered in LDS via async loads.
__global__ void __launch_bounds__(256)
graph_gemm_wmma_k(const u16* __restrict__ Abf, const u16* __restrict__ Xpack,
                  u16* __restrict__ G) {
  __shared__ u16 lds[2][8 * 2048];                // 2 x 32KB chunks
  int tid = threadIdx.x;
  int wg = blockIdx.x;                            // 8b * 7ct * 8mg = 448
  int b  = wg / (7 * 8);
  int r0 = wg % (7 * 8);
  int ct = r0 / 8;
  int mg = r0 % 8;
  int wv = tid >> 5, lane = tid & 31;
  int mbase = (mg * 8 + wv) * 32;                 // 32 rows per wave
  int kb = (lane >> 4) << 3;
  const u16* arow0 = Abf + (size_t)b * Nn * Nn + (size_t)(mbase + (lane & 15)) * Nn + kb;
  const u16* arow1 = arow0 + (size_t)16 * Nn;
  const u16* xbase = Xpack + (size_t)b * XPK_BATCH + (size_t)(ct * 4) * 512;

  // stage chunk c (k-tiles c*8..c*8+7): 8 rows x 2048 halves, 1 b128/thread/row
#define STAGE(c, buf)                                                         \
  {                                                                           \
    _Pragma("unroll")                                                         \
    for (int j = 0; j < 8; ++j)                                               \
      cp_g2s_b128(&lds[buf][j * 2048 + tid * 8],                              \
                  xbase + (size_t)(((c) * 8 + j) * 28) * 512 + tid * 8);      \
  }

  f32v8 acc[8] = {};                              // [0..3]=rows 0-15, [4..7]=rows 16-31
  bf16v16 a0 = load_frag_a(arow0);                // kt = 0, issued early
  bf16v16 a1 = load_frag_a(arow1);
  STAGE(0, 0);
  for (int c = 0; c < 8; ++c) {
    wait_cp_all();
    __syncthreads();                              // chunk c visible; prev reads done
    if (c + 1 < 8) STAGE(c + 1, (c + 1) & 1);     // overlap next chunk with compute
    const u16* bl = lds[c & 1] + lane * 16;
#pragma unroll
    for (int j = 0; j < 8; ++j) {
      bf16v16 ac0 = a0, ac1 = a1;
      int ktn = c * 8 + j + 1;
      if (ktn < 64) {
        a0 = load_frag_a(arow0 + ktn * 32);
        a1 = load_frag_a(arow1 + ktn * 32);
      }
#pragma unroll
      for (int nt = 0; nt < 4; ++nt) {
        bf16v16 bf = load_frag_b(bl + (j * 4 + nt) * 512);
        acc[nt]     = wmma_bf16(ac0, bf, acc[nt]);
        acc[4 + nt] = wmma_bf16(ac1, bf, acc[4 + nt]);
      }
    }
  }
#undef STAGE
  int row0 = mbase + ((lane >> 4) << 3);
  int col0 = ct * 64 + (lane & 15);
  u16* Gb = G + (size_t)b * Nn * 448;
#pragma unroll
  for (int nt = 0; nt < 4; ++nt)
#pragma unroll
    for (int r = 0; r < 8; ++r) {
      Gb[(size_t)(row0 + r) * 448 + col0 + nt * 16]      = f2bf(acc[nt][r]);
      Gb[(size_t)(row0 + 16 + r) * 448 + col0 + nt * 16] = f2bf(acc[4 + nt][r]);
    }
}

// relu(G @ Wg + bg): M = B*N*7 rows, K = 64, N = 64. All 8 B fragments are
// held in registers (64 VGPRs), loaded once before the (tiny) K loop.
__global__ void __launch_bounds__(256)
gemm_relu_wmma_k(const u16* __restrict__ G, const u16* __restrict__ WgPack,
                 const float* __restrict__ bg, u16* __restrict__ H, int nWaves) {
  int wave = (blockIdx.x * blockDim.x + threadIdx.x) >> 5;
  if (wave >= nWaves) return;
  int m0 = wave * 16;
  int lane = threadIdx.x & 31;
  int kb = (lane >> 4) << 3;
  const u16* arow = G + (size_t)(m0 + (lane & 15)) * 64 + kb;
  const u16* wl = WgPack + lane * 16;

  bf16v16 bfr[8];
#pragma unroll
  for (int i = 0; i < 8; ++i) bfr[i] = load_frag_b(wl + i * 512);
  bf16v16 a0 = load_frag_a(arow);
  bf16v16 a1 = load_frag_a(arow + 32);

  f32v8 acc[4] = {};
#pragma unroll
  for (int nt = 0; nt < 4; ++nt) {
    acc[nt] = wmma_bf16(a0, bfr[nt], acc[nt]);
    acc[nt] = wmma_bf16(a1, bfr[4 + nt], acc[nt]);
  }
  int row0 = m0 + ((lane >> 4) << 3);
  int colb = (lane & 15);
#pragma unroll
  for (int nt = 0; nt < 4; ++nt) {
    int col = nt * 16 + colb;
    float vb = bg[col];
#pragma unroll
    for (int r = 0; r < 8; ++r) {
      float v = acc[nt][r] + vb;
      H[(size_t)(row0 + r) * 64 + col] = f2bf(v > 0.f ? v : 0.f);
    }
  }
}

// Final projection: out[b,n] = y3[b,n,:] . Wout + bout  (64 -> 1)
__global__ void out_proj_k(const u16* __restrict__ y3, const float* __restrict__ Wout,
                           const float* __restrict__ bout, float* __restrict__ out, int n) {
  int i = blockIdx.x * blockDim.x + threadIdx.x;
  if (i >= n) return;
  float s = 0.f;
#pragma unroll
  for (int c = 0; c < 64; ++c) s += bf2f(y3[(size_t)i * 64 + c]) * Wout[c];
  out[i] = s + bout[0];
}

// ---------------- workspace layout ----------------
constexpr size_t al256(size_t x) { return (x + 255) & ~(size_t)255; }
constexpr size_t SZ_XBF  = (size_t)Bn * Nn * Ln * CIN0 * 2;   // 33.5 MB
constexpr size_t SZ_DEG  = (size_t)Bn * Nn * 4;
constexpr size_t SZ_ACNT = (size_t)Bn * Nn * Nn * 4;          // 134 MB (fp32 counts)
constexpr size_t SZ_ABF  = (size_t)Bn * Nn * Nn * 2;          // 67 MB  (bf16 A_hat)
constexpr size_t SZ_WP1  = 256 * 64 * 2;
constexpr size_t SZ_WG   = 64 * 64 * 2;
constexpr size_t SZ_WP2  = 128 * 64 * 2;
constexpr size_t SZ_WP3  = 384 * 64 * 2;
constexpr size_t SZ_XPK  = (size_t)Bn * XPK_BATCH * 2;        // 14.7 MB
constexpr size_t SZ_G1   = (size_t)Bn * Nn * L1 * Cc * 2;
constexpr size_t SZ_H1   = SZ_G1;
constexpr size_t SZ_Y2   = (size_t)Bn * Nn * L2c * Cc * 2;

constexpr size_t OFF_XBF  = 0;
constexpr size_t OFF_DEG  = OFF_XBF + al256(SZ_XBF);
constexpr size_t OFF_ACNT = OFF_DEG + al256(SZ_DEG);
constexpr size_t OFF_ABF  = OFF_ACNT + al256(SZ_ACNT);
constexpr size_t OFF_WP1  = OFF_ABF + al256(SZ_ABF);
constexpr size_t OFF_WQ1  = OFF_WP1 + al256(SZ_WP1);
constexpr size_t OFF_WG   = OFF_WQ1 + al256(SZ_WP1);
constexpr size_t OFF_WP2  = OFF_WG  + al256(SZ_WG);
constexpr size_t OFF_WQ2  = OFF_WP2 + al256(SZ_WP2);
constexpr size_t OFF_WP3  = OFF_WQ2 + al256(SZ_WP2);
constexpr size_t OFF_WQ3  = OFF_WP3 + al256(SZ_WP3);
constexpr size_t OFF_XPK  = OFF_WQ3 + al256(SZ_WP3);
// Overlay post-adjacency activations onto the (dead) fp32 count buffer:
constexpr size_t OFF_G1   = OFF_ACNT;
constexpr size_t OFF_H1   = OFF_G1 + al256(SZ_G1);
constexpr size_t OFF_Y2   = OFF_H1 + al256(SZ_H1);
constexpr size_t OFF_Y3   = OFF_Y2 + al256(SZ_Y2);

static inline int cdiv(long long a, long long b) { return (int)((a + b - 1) / b); }

extern "C" void kernel_launch(void* const* d_in, const int* in_sizes, int n_in,
                              void* d_out, int out_size, void* d_ws, size_t ws_size,
                              hipStream_t stream) {
  const float* prop = (const float*)d_in[0];
  const float* ann  = (const float*)d_in[1];
  const float* Ain  = (const float*)d_in[2];
  const float* Wp1  = (const float*)d_in[3];
  const float* bp1  = (const float*)d_in[4];
  const float* Wq1  = (const float*)d_in[5];
  const float* bq1  = (const float*)d_in[6];
  const float* Wg   = (const float*)d_in[7];
  const float* bg   = (const float*)d_in[8];
  const float* Wp2  = (const float*)d_in[9];
  const float* bp2  = (const float*)d_in[10];
  const float* Wq2  = (const float*)d_in[11];
  const float* bq2  = (const float*)d_in[12];
  const float* Wp3  = (const float*)d_in[13];
  const float* bp3  = (const float*)d_in[14];
  const float* Wq3  = (const float*)d_in[15];
  const float* bq3  = (const float*)d_in[16];
  const float* Wout = (const float*)d_in[17];
  const float* bout = (const float*)d_in[18];
  float* out = (float*)d_out;

  char* ws = (char*)d_ws;
  u16*   xbf   = (u16*)(ws + OFF_XBF);
  float* deg   = (float*)(ws + OFF_DEG);
  float* Acnt  = (float*)(ws + OFF_ACNT);
  u16*   Abf   = (u16*)(ws + OFF_ABF);
  u16*   wp1p  = (u16*)(ws + OFF_WP1);
  u16*   wq1p  = (u16*)(ws + OFF_WQ1);
  u16*   wgp   = (u16*)(ws + OFF_WG);
  u16*   wp2p  = (u16*)(ws + OFF_WP2);
  u16*   wq2p  = (u16*)(ws + OFF_WQ2);
  u16*   wp3p  = (u16*)(ws + OFF_WP3);
  u16*   wq3p  = (u16*)(ws + OFF_WQ3);
  u16*   xpk   = (u16*)(ws + OFF_XPK);
  u16*   g1    = (u16*)(ws + OFF_G1);
  u16*   h1    = (u16*)(ws + OFF_H1);
  u16*   y2    = (u16*)(ws + OFF_Y2);
  u16*   y3    = (u16*)(ws + OFF_Y3);

  const int T = 256;

  // --- adjacency normalization path ---
  deg_init_k<<<cdiv(Bn * Nn, T), T, 0, stream>>>(deg, Bn * Nn);
  deg_scatter_k<<<cdiv((long long)Bn * NNZn, T), T, 0, stream>>>(Ain, deg);
  dinv_k<<<cdiv(Bn * Nn, T), T, 0, stream>>>(deg, Bn * Nn);
  zero_f4_k<<<cdiv((long long)Bn * Nn * Nn / 4, T), T, 0, stream>>>((float4*)Acnt,
                                                                    (size_t)Bn * Nn * Nn / 4);
  adj_scatter_k<<<cdiv((long long)Bn * NNZn, T), T, 0, stream>>>(Ain, Acnt);
  adj_norm_k<<<cdiv((long long)Bn * Nn * Nn / 4, T), T, 0, stream>>>(Acnt, deg, Abf);

  // --- input concat -> bf16; weight packs ---
  concat_bf16_k<<<cdiv((long long)Bn * Nn * Ln * CIN0 / 4, T), T, 0, stream>>>(prop, ann, xbf);
  pack_b_f32_k<<<cdiv(256 * 64, T), T, 0, stream>>>(Wp1, wp1p, 256, 64);
  pack_b_f32_k<<<cdiv(256 * 64, T), T, 0, stream>>>(Wq1, wq1p, 256, 64);
  pack_b_f32_k<<<cdiv(64 * 64, T), T, 0, stream>>>(Wg, wgp, 64, 64);
  pack_b_f32_k<<<cdiv(128 * 64, T), T, 0, stream>>>(Wp2, wp2p, 128, 64);
  pack_b_f32_k<<<cdiv(128 * 64, T), T, 0, stream>>>(Wq2, wq2p, 128, 64);
  pack_b_f32_k<<<cdiv(384 * 64, T), T, 0, stream>>>(Wp3, wp3p, 384, 64);
  pack_b_f32_k<<<cdiv(384 * 64, T), T, 0, stream>>>(Wq3, wq3p, 384, 64);

  // --- conv1 (fused K=256 -> 8 k-tiles), L 8->7; writes B-fragment-packed X ---
  {
    int nWaves = (Bn * Nn / 16) * L1;   // 7168
    conv_glu_wmma_k<8><<<cdiv((long long)nWaves * 32, T), T, 256 * 256, stream>>>(
        xbf, Ln * CIN0, CIN0, wp1p, wq1p, bp1, bq1, xpk, L1 * Cc, L1, 1, nWaves);
  }
  // --- graph conv: C = A_hat x X (dominant GEMM, 32x64 wave tiles) ---
  graph_gemm_wmma_k<<<Bn * 7 * 8, T, 0, stream>>>(Abf, xpk, g1);

  // --- relu(G @ Wg + bg) ---
  {
    int nWaves = (Bn * Nn * L1) / 16;   // 7168
    gemm_relu_wmma_k<<<cdiv((long long)nWaves * 32, T), T, 0, stream>>>(g1, wgp, bg, h1, nWaves);
  }
  // --- conv2 (fused K=128 -> 4 k-tiles), L 7->6 ---
  {
    int nWaves = (Bn * Nn / 16) * L2c;  // 6144
    conv_glu_wmma_k<4><<<cdiv((long long)nWaves * 32, T), T, 128 * 256, stream>>>(
        h1, L1 * Cc, Cc, wp2p, wq2p, bp2, bq2, y2, L2c * Cc, L2c, 0, nWaves);
  }
  // --- conv3 (fused K=384 -> 12 k-tiles), L 6->1 ---
  {
    int nWaves = (Bn * Nn / 16);        // 1024
    conv_glu_wmma_k<12><<<cdiv((long long)nWaves * 32, T), T, 384 * 256, stream>>>(
        y2, L2c * Cc, Cc, wp3p, wq3p, bp3, bq3, y3, Cc, 1, 0, nWaves);
  }
  // --- final 64->1 projection ---
  out_proj_k<<<cdiv(Bn * Nn, T), T, 0, stream>>>(y3, Wout, bout, out, Bn * Nn);

  (void)in_sizes; (void)n_in; (void)out_size; (void)ws_size;
}
